// CausalSelfAttention_65094524338586
// MI455X (gfx1250) — compile-verified, matrix operand order checked by reading, not run on hardware
//
#include <hip/hip_runtime.h>
#include <stdint.h>

// ---------------------------------------------------------------------------
// Causal self-attention forward for MI455X (gfx1250, wave32, WMMA).
// Pipeline: [x @ Wqkv] (WMMA) -> RoPE+repack -> flash attention (WMMA, with
//           GLOBAL_LOAD_ASYNC_TO_LDS_B128 K-tile staging) -> [att @ Wproj].
// All matmuls use v_wmma_f32_16x16x32_bf16 (bf16 in, f32 accumulate).
// ---------------------------------------------------------------------------

#define B_  2
#define T_  2048
#define D_  1024
#define H_  16
#define HD_ 64

typedef __attribute__((ext_vector_type(16))) __bf16 v16bf;
typedef __attribute__((ext_vector_type(8)))  float  v8f;

union AFrag { v16bf v; uint32_t u[8]; };
union CFrag { v8f   v; float    f[8]; };

// round-to-nearest-even float -> bf16 raw bits
__device__ __forceinline__ unsigned short f2bf_bits(float f) {
    union { float f; uint32_t u; } v; v.f = f;
    uint32_t r = v.u + 0x7FFFu + ((v.u >> 16) & 1u);
    return (unsigned short)(r >> 16);
}

// K-index of the (even) bf16 pair held by VGPR p of a 16-bit A/B fragment.
// ISA 16-bit 16x32 layout: lanes 0-15 hold k {0..7,16..23}, lanes 16-31 hold
// k {8..15,24..31}; VGPR p packs two consecutive k.
__device__ __forceinline__ int kmap(int p, int hl) {
    return ((p < 4) ? (p << 1) : (16 + ((p - 4) << 1))) + (hl << 3);
}

// Async 16-byte copy global -> LDS (per-lane addresses), tracked by ASYNCcnt.
__device__ __forceinline__ void async_ld_b128(uint32_t lds_off, const void* gptr) {
    asm volatile("global_load_async_to_lds_b128 %0, %1, off"
                 :: "v"(lds_off), "v"((uint64_t)(uintptr_t)gptr)
                 : "memory");
}
__device__ __forceinline__ void wait_async0() {
    asm volatile("s_wait_asynccnt 0x0" ::: "memory");
}

// ---------------------------------------------------------------------------
// Generic GEMM: C[M,N] = A[M,K] @ B[K,N], A/B fp32 row-major converted to bf16
// in LDS, C fp32. Block = 256 threads (8 waves), tile 128x128, K-step 32.
// ---------------------------------------------------------------------------
#define TM 128
#define TN 128
#define TK 32
#define APITCH 34   // ushort pitch (even -> u32-aligned; odd dword stride)
#define BPITCH 34

__global__ __launch_bounds__(256) void gemm_bf16_wmma(
    const float* __restrict__ A, const float* __restrict__ Bm,
    float* __restrict__ C, int M, int N, int K)
{
    __shared__ unsigned short As[TM * APITCH];  // [m][k]
    __shared__ unsigned short Bs[TN * BPITCH];  // transposed: [n][k]

    const int tid  = threadIdx.x;
    const int lane = tid & 31;
    const int w    = tid >> 5;
    const int wm   = w & 3;       // 4 waves along M (32 rows each)
    const int wn   = w >> 2;      // 2 waves along N (64 cols each)
    const int hl   = lane >> 4;
    const int l15  = lane & 15;
    const int m_blk = blockIdx.y * TM;
    const int n_blk = blockIdx.x * TN;

    CFrag acc[2][4];
#pragma unroll
    for (int mi = 0; mi < 2; ++mi)
#pragma unroll
        for (int ni = 0; ni < 4; ++ni)
            acc[mi][ni].v = (v8f){0.f,0.f,0.f,0.f,0.f,0.f,0.f,0.f};

    for (int kk = 0; kk < K; kk += TK) {
        // stage A tile (128x32): 8 rows per pass, 32 lanes cover one k-row
        {
            const int kcol = tid & 31;
            const int r0   = tid >> 5;
#pragma unroll
            for (int r = r0; r < TM; r += 8)
                As[r * APITCH + kcol] =
                    f2bf_bits(A[(size_t)(m_blk + r) * K + kk + kcol]);
        }
        // stage B tile (32x128) transposed into [n][k]
        {
            const int ncol = tid & 127;
            const int kr0  = tid >> 7;
#pragma unroll
            for (int kr = kr0; kr < TK; kr += 2)
                Bs[ncol * BPITCH + kr] =
                    f2bf_bits(Bm[(size_t)(kk + kr) * N + n_blk + ncol]);
        }
        __syncthreads();

        AFrag a[2], b[4];
#pragma unroll
        for (int mi = 0; mi < 2; ++mi) {
            const int row = wm * 32 + mi * 16 + l15;
            const uint32_t* src = (const uint32_t*)(As + row * APITCH);
#pragma unroll
            for (int p = 0; p < 8; ++p) a[mi].u[p] = src[kmap(p, hl) >> 1];
        }
#pragma unroll
        for (int ni = 0; ni < 4; ++ni) {
            const int col = wn * 64 + ni * 16 + l15;
            const uint32_t* src = (const uint32_t*)(Bs + col * BPITCH);
#pragma unroll
            for (int p = 0; p < 8; ++p) b[ni].u[p] = src[kmap(p, hl) >> 1];
        }
#pragma unroll
        for (int mi = 0; mi < 2; ++mi)
#pragma unroll
            for (int ni = 0; ni < 4; ++ni)
                acc[mi][ni].v = __builtin_amdgcn_wmma_f32_16x16x32_bf16(
                    false, a[mi].v, false, b[ni].v,
                    (short)0, acc[mi][ni].v, false, false);
        __syncthreads();
    }

    // C layout: lane holds col = l15, rows v + 8*hl per VGPR v
#pragma unroll
    for (int mi = 0; mi < 2; ++mi)
#pragma unroll
        for (int ni = 0; ni < 4; ++ni)
#pragma unroll
            for (int v = 0; v < 8; ++v) {
                const int row = m_blk + wm * 32 + mi * 16 + v + 8 * hl;
                const int col = n_blk + wn * 64 + ni * 16 + l15;
                C[(size_t)row * N + col] = acc[mi][ni].f[v];
            }
}

// ---------------------------------------------------------------------------
// RoPE + repack: qkv fp32 (B,T,3,H,HD) -> Q,K rotated + V as bf16 (B,H,T,HD)
// ---------------------------------------------------------------------------
__global__ __launch_bounds__(256) void rope_pack(
    const float* __restrict__ qkv,
    unsigned short* __restrict__ Qb,
    unsigned short* __restrict__ Kb,
    unsigned short* __restrict__ Vb)
{
    const int idx = blockIdx.x * blockDim.x + threadIdx.x;  // 2^22 threads
    const int d = idx & (HD_ - 1);
    const int h = (idx >> 6) & (H_ - 1);
    const int t = (idx >> 10) & (T_ - 1);
    const int b = idx >> 21;

    const size_t sStride = (size_t)H_ * HD_;                       // 1024
    const size_t base = ((((size_t)b * T_ + t) * 3) * H_ + h) * HD_ + d;
    const float q  = qkv[base];
    const float k  = qkv[base + sStride];
    const float vv = qkv[base + 2 * sStride];

    const int   dp   = (d < 32) ? d + 32 : d - 32;
    const float qp   = qkv[base + (dp - d)];
    const float kp   = qkv[base + sStride + (dp - d)];
    const float rq   = (d < 32) ? -qp : qp;
    const float rk   = (d < 32) ? -kp : kp;

    const int   fi  = d & 31;
    const float inv = __powf(10000.0f, -(float)fi * (1.0f / 32.0f));
    const float ang = (float)t * inv;
    float s, c;
    __sincosf(ang, &s, &c);

    const size_t o = (((size_t)b * H_ + h) * T_ + t) * HD_ + d;
    Qb[o] = f2bf_bits(q * c + rq * s);
    Kb[o] = f2bf_bits(k * c + rk * s);
    Vb[o] = f2bf_bits(vv);
}

// ---------------------------------------------------------------------------
// Flash attention: 1 workgroup = (b, h, 64-query block); 4 waves x 16 queries.
// Streaming softmax over 32-key tiles; S and P*V via WMMA.
// K tile staged by GLOBAL_LOAD_ASYNC_TO_LDS_B128; V loaded b128 + transposed.
// ---------------------------------------------------------------------------
#define QBLK 64
#define KBLK 32
#define KPITCH 72   // K tile [key][d]; 144B rows: 16B-aligned for async b128,
                    // 36-dword lane stride -> 16 distinct banks on frag reads
#define VPITCH 34   // V tile transposed [d][key]
#define PPITCH 34   // per-wave P buffer [q][key]

__global__ __launch_bounds__(128) void flash_attn(
    const unsigned short* __restrict__ Qb,
    const unsigned short* __restrict__ Kb,
    const unsigned short* __restrict__ Vb,
    const unsigned char*  __restrict__ kpm,
    float* __restrict__ att)
{
    __shared__ unsigned short Kt[KBLK * KPITCH];
    __shared__ unsigned short Vt[HD_ * VPITCH];
    __shared__ unsigned short Pb[4][16 * PPITCH];

    const int tid  = threadIdx.x;
    const int lane = tid & 31;
    const int w    = tid >> 5;
    const int hl   = lane >> 4;
    const int l15  = lane & 15;

    const int qtile = blockIdx.x & 31;          // T/64 = 32
    const int h     = (blockIdx.x >> 5) & 15;
    const int b     = blockIdx.x >> 9;
    const int q0    = qtile * QBLK;

    const size_t headOff = ((size_t)b * H_ + h) * (size_t)T_ * HD_;
    const unsigned short* Qh = Qb + headOff;
    const unsigned short* Kh = Kb + headOff;
    const unsigned short* Vh = Vb + headOff;

    // Q A-fragments for this wave's 16 rows (held for the whole loop)
    AFrag qa[2];
    {
        const int qrow = q0 + w * 16 + l15;
        const unsigned short* qp = Qh + (size_t)qrow * HD_;
#pragma unroll
        for (int f = 0; f < 2; ++f)
#pragma unroll
            for (int p = 0; p < 8; ++p)
                qa[f].u[p] = *(const uint32_t*)(qp + kmap(p, hl) + f * 32);
    }

    CFrag O[4];
#pragma unroll
    for (int f = 0; f < 4; ++f)
        O[f].v = (v8f){0.f,0.f,0.f,0.f,0.f,0.f,0.f,0.f};
    float mstat[8], lstat[8];
#pragma unroll
    for (int v = 0; v < 8; ++v) { mstat[v] = -3.0e38f; lstat[v] = 0.0f; }

    // staging decomposition: 128 threads cover 32 rows x 4 chunks of 16 elems
    const int sr = tid >> 2;            // key row 0..31
    const int sd = (tid & 3) << 4;      // d group 0,16,32,48

    const int nkt = (q0 + QBLK) / KBLK;   // causal: keys <= q0+63
    for (int kt = 0; kt < nkt; ++kt) {
        const int kbase = kt * KBLK;
        __syncthreads();                       // guard prior-iter LDS reads
        {
            const unsigned short* kRow = Kh + (size_t)(kbase + sr) * HD_ + sd;
            const unsigned short* vRow = Vh + (size_t)(kbase + sr) * HD_ + sd;
            // K: two async 16B global->LDS copies into padded row-major tile
            const uint32_t ldsK =
                (uint32_t)(uintptr_t)(Kt + sr * KPITCH + sd);
            async_ld_b128(ldsK,      kRow);
            async_ld_b128(ldsK + 16, kRow + 8);
            // V: two b128 register loads + transposed LDS stores [d][key]
            union { uint4 q[2]; unsigned short e[16]; } vv;
            vv.q[0] = *(const uint4*)(vRow);
            vv.q[1] = *(const uint4*)(vRow + 8);
#pragma unroll
            for (int e = 0; e < 16; ++e)
                Vt[(sd + e) * VPITCH + sr] = vv.e[e];
        }
        wait_async0();                         // our async copies landed
        __syncthreads();                       // everyone's did

        // S = Q . K^T  (two 16-key chunks, K-dim 64 = 2 chained WMMAs)
        CFrag S[2];
#pragma unroll
        for (int c = 0; c < 2; ++c) {
            const unsigned short* kp = Kt + (c * 16 + l15) * KPITCH;
            AFrag kb0, kb1;
#pragma unroll
            for (int p = 0; p < 8; ++p) {
                kb0.u[p] = *(const uint32_t*)(kp + kmap(p, hl));
                kb1.u[p] = *(const uint32_t*)(kp + kmap(p, hl) + 32);
            }
            v8f z = (v8f){0.f,0.f,0.f,0.f,0.f,0.f,0.f,0.f};
            S[c].v = __builtin_amdgcn_wmma_f32_16x16x32_bf16(
                false, qa[0].v, false, kb0.v, (short)0, z, false, false);
            S[c].v = __builtin_amdgcn_wmma_f32_16x16x32_bf16(
                false, qa[1].v, false, kb1.v, (short)0, S[c].v, false, false);

            // scale + causal + padding mask (C layout)
            const int j = kbase + c * 16 + l15;
            const unsigned char pm = kpm[b * T_ + j];
#pragma unroll
            for (int v = 0; v < 8; ++v) {
                const int qi = q0 + w * 16 + v + 8 * hl;
                float sv = S[c].f[v] * 0.125f;    // 1/sqrt(64)
                if (j > qi || !pm) sv = -3.0e38f;
                S[c].f[v] = sv;
            }
        }

        // online softmax update (row reduction across the 16-lane half)
#pragma unroll
        for (int v = 0; v < 8; ++v) {
            float rm = fmaxf(S[0].f[v], S[1].f[v]);
#pragma unroll
            for (int off = 8; off >= 1; off >>= 1)
                rm = fmaxf(rm, __shfl_xor(rm, off, 32));
            const float nm = fmaxf(mstat[v], rm);
            const float sc = __expf(mstat[v] - nm);
            const float p0 = __expf(S[0].f[v] - nm);
            const float p1 = __expf(S[1].f[v] - nm);
            S[0].f[v] = p0; S[1].f[v] = p1;
            float rs = p0 + p1;
#pragma unroll
            for (int off = 8; off >= 1; off >>= 1)
                rs += __shfl_xor(rs, off, 32);
            lstat[v] = lstat[v] * sc + rs;
            mstat[v] = nm;
#pragma unroll
            for (int f = 0; f < 4; ++f) O[f].f[v] *= sc;
        }

        // P: C layout -> LDS -> A layout
        {
            unsigned short* pw = Pb[w];
#pragma unroll
            for (int c = 0; c < 2; ++c)
#pragma unroll
                for (int v = 0; v < 8; ++v)
                    pw[(v + 8 * hl) * PPITCH + c * 16 + l15] =
                        f2bf_bits(S[c].f[v]);
        }
        __syncthreads();

        AFrag pa;
        {
            const unsigned short* pr = Pb[w] + l15 * PPITCH;
#pragma unroll
            for (int p = 0; p < 8; ++p)
                pa.u[p] = *(const uint32_t*)(pr + kmap(p, hl));
        }
        // O += P . V  (4 d-chunks of 16)
#pragma unroll
        for (int f = 0; f < 4; ++f) {
            const unsigned short* vr = Vt + (f * 16 + l15) * VPITCH;
            AFrag vb;
#pragma unroll
            for (int p = 0; p < 8; ++p)
                vb.u[p] = *(const uint32_t*)(vr + kmap(p, hl));
            O[f].v = __builtin_amdgcn_wmma_f32_16x16x32_bf16(
                false, pa.v, false, vb.v, (short)0, O[f].v, false, false);
        }
    }

    // normalize and write to (B,T,D) fp32 for the projection GEMM
#pragma unroll
    for (int v = 0; v < 8; ++v) {
        const float il = 1.0f / lstat[v];
        const int t = q0 + w * 16 + v + 8 * hl;
#pragma unroll
        for (int f = 0; f < 4; ++f)
            att[((size_t)b * T_ + t) * D_ + h * HD_ + f * 16 + l15] =
                O[f].f[v] * il;
    }
}

// ---------------------------------------------------------------------------
extern "C" void kernel_launch(void* const* d_in, const int* in_sizes, int n_in,
                              void* d_out, int out_size, void* d_ws, size_t ws_size,
                              hipStream_t stream)
{
    (void)in_sizes; (void)n_in; (void)out_size; (void)ws_size;
    const float*         x     = (const float*)d_in[0];
    const float*         Wqkv  = (const float*)d_in[1];
    const float*         Wproj = (const float*)d_in[2];
    const unsigned char* kpm   = (const unsigned char*)d_in[3];

    // workspace carve-out (~88 MB total)
    char* ws = (char*)d_ws;
    size_t off = 0;
    float* qkv = (float*)(ws + off);          off += (size_t)B_ * T_ * 3 * D_ * 4;
    unsigned short* Qb = (unsigned short*)(ws + off); off += (size_t)B_ * T_ * D_ * 2;
    unsigned short* Kb = (unsigned short*)(ws + off); off += (size_t)B_ * T_ * D_ * 2;
    unsigned short* Vb = (unsigned short*)(ws + off); off += (size_t)B_ * T_ * D_ * 2;
    float* att = (float*)(ws + off);          off += (size_t)B_ * T_ * D_ * 4;

    const int M = B_ * T_;                    // 4096

    // 1) QKV projection: (4096 x 1024) @ (1024 x 3072)
    gemm_bf16_wmma<<<dim3((3 * D_) / TN, M / TM), 256, 0, stream>>>(
        x, Wqkv, qkv, M, 3 * D_, D_);

    // 2) RoPE + repack to (B,H,T,HD) bf16
    rope_pack<<<(B_ * T_ * D_) / 256, 256, 0, stream>>>(qkv, Qb, Kb, Vb);

    // 3) flash attention
    flash_attn<<<B_ * H_ * (T_ / QBLK), 128, 0, stream>>>(Qb, Kb, Vb, kpm, att);

    // 4) output projection: (4096 x 1024) @ (1024 x 1024) -> d_out
    gemm_bf16_wmma<<<dim3(D_ / TN, M / TM), 256, 0, stream>>>(
        att, Wproj, (float*)d_out, M, D_, D_);
}
